// Qwen3NextSparseMoeBlock_26654567038979
// MI455X (gfx1250) — matrix operand order repaired and checked by reading, not compiled, not run on hardware
//
#include <hip/hip_runtime.h>

#define T_TOK 1024
#define HID   2048
#define NEXP  64
#define IDIM  512
#define ISH   1024
#define TOPK  8
#define CAP   256

typedef __bf16 bf16;
typedef __attribute__((ext_vector_type(16))) __bf16 v16bf;
typedef __attribute__((ext_vector_type(4)))  __bf16 v4bf;
typedef __attribute__((ext_vector_type(8)))  float  v8f;
typedef __attribute__((ext_vector_type(2)))  int    v2i;

#if __has_builtin(__builtin_amdgcn_global_load_async_to_lds_b64)
#define USE_ASYNC_LDS 1
#else
#define USE_ASYNC_LDS 0
#endif

__device__ __forceinline__ void wait_async0() {
#if __has_builtin(__builtin_amdgcn_s_wait_asynccnt)
  __builtin_amdgcn_s_wait_asynccnt(0);
#else
  asm volatile("s_wait_asynccnt 0x0" ::: "memory");
#endif
}

#if USE_ASYNC_LDS
// b64 async copy: builtin is typed (v2i __global*, v2i __shared*, imm, imm)
__device__ __forceinline__ void async_copy_b64(const bf16* gsrc, bf16* ldst) {
  __builtin_amdgcn_global_load_async_to_lds_b64(
      (__attribute__((address_space(1))) v2i*)gsrc,
      (__attribute__((address_space(3))) v2i*)ldst, 0, 0);
}
#endif

// ---------------------------------------------------------------------------
// WMMA fragment helpers (wave32, per CDNA5 ISA 7.12.2 layouts)
// A (16x32 bf16, row-major LDS tile, ld=32):
//   lane L<16 : M=L, elems 0..7 = K 0..7,  elems 8..15 = K 16..23
//   lane L>=16: M=L-16, elems 0..7 = K 8..15, elems 8..15 = K 24..31
// B (32x16 bf16) stored transposed in LDS as [n][k] (16 x 32):
//   lane L<16 : N=L, elems 0..7 = K 0..7, 8..15 = K 16..23 ; lane L>=16: N=L-16, elems 0..7 = K 8..15, 8..15 = K 24..31
// C/D (16x16 f32): elem v of lane L -> M = v + 8*(L>=16), N = L&15
// ---------------------------------------------------------------------------
__device__ __forceinline__ v16bf load_a_frag(const bf16* ldsA) {
  const int lane = threadIdx.x & 31;
  const int row  = lane & 15;
  const int kh   = (lane >> 4) << 3;       // 0 or 8
  const bf16* p = ldsA + row * 32 + kh;
  v16bf a;
#pragma unroll
  for (int i = 0; i < 8; ++i) a[i] = p[i];
#pragma unroll
  for (int i = 0; i < 8; ++i) a[8 + i] = p[16 + i];
  return a;
}

__device__ __forceinline__ v16bf load_b_frag(const bf16* ldsBT) {
  const int lane = threadIdx.x & 31;
  const int col  = lane & 15;
  const int kb   = (lane >> 4) << 3;       // 0 or 8
  const bf16* p = ldsBT + col * 32 + kb;
  v16bf b;
#pragma unroll
  for (int i = 0; i < 8; ++i) { b[i] = p[i]; b[8 + i] = p[16 + i]; }
  return b;
}

__device__ __forceinline__ v8f wmma_bf16(v16bf a, v16bf b, v8f c) {
  // (neg_a, A, neg_b, B, c_mod, C, reuse_a, reuse_b)
  return __builtin_amdgcn_wmma_f32_16x16x32_bf16(false, a, false, b, (short)0, c,
                                                 false, false);
}

__device__ __forceinline__ void zero8(v8f& v) {
#pragma unroll
  for (int i = 0; i < 8; ++i) v[i] = 0.0f;
}

// ---------------------------------------------------------------------------
// Router: per-token logits over 64 experts, top-8, renormalized softmax.
// ---------------------------------------------------------------------------
__global__ __launch_bounds__(64)
void moe_router(const float* __restrict__ x, const float* __restrict__ gate_w,
                int* __restrict__ ridx, float* __restrict__ rw) {
  __shared__ float lg[NEXP];
  const int t = blockIdx.x, e = threadIdx.x;
  const float* xr = x + (size_t)t * HID;
  const float* g  = gate_w + (size_t)e * HID;
  float acc = 0.0f;
  for (int i = 0; i < HID; i += 4) {
    float4 a = *(const float4*)(xr + i);
    float4 b = *(const float4*)(g + i);
    acc += a.x * b.x + a.y * b.y + a.z * b.z + a.w * b.w;
  }
  lg[e] = acc;
  __syncthreads();
  if (e == 0) {
    float vals[TOPK]; int idx[TOPK];
#pragma unroll
    for (int k = 0; k < TOPK; ++k) {
      float best = -3.4e38f; int bi = 0;
      for (int j = 0; j < NEXP; ++j) {
        float v = lg[j];
        if (v > best) { best = v; bi = j; }   // strict > => lowest index on tie
      }
      vals[k] = best; idx[k] = bi; lg[bi] = -3.4e38f;
    }
    const float m = vals[0];
    float s = 0.0f, w[TOPK];
#pragma unroll
    for (int k = 0; k < TOPK; ++k) { w[k] = __expf(vals[k] - m); s += w[k]; }
    const float inv = 1.0f / s;
#pragma unroll
    for (int k = 0; k < TOPK; ++k) {
      ridx[t * TOPK + k] = idx[k];
      rw[t * TOPK + k]   = w[k] * inv;
    }
  }
}

// ---------------------------------------------------------------------------
// Dispatch: one wave per expert scans all T*TOPK flat entries in order and
// assigns stable ranks via ballot prefix-popcount (matches JAX stable sort).
// ---------------------------------------------------------------------------
__global__ __launch_bounds__(32)
void moe_dispatch(const int* __restrict__ ridx, int* __restrict__ etok,
                  int* __restrict__ epos) {
  const int e = blockIdx.x;
  const int lane = threadIdx.x;
  for (int p = lane; p < CAP; p += 32) etok[e * CAP + p] = 0;  // safe row id
  int base = 0;
  for (int i0 = 0; i0 < T_TOK * TOPK; i0 += 32) {
    const int i = i0 + lane;
    const bool m = (ridx[i] == e);
    const unsigned mask = (unsigned)__ballot(m);
    const int pre = __popc(mask & ((1u << lane) - 1u));
    if (m) {
      const int pos = base + pre;
      if (pos < CAP) { etok[e * CAP + pos] = i >> 3; epos[i] = pos; }
      else           { epos[i] = -1; }
    }
    base += __popc(mask);
  }
}

// ---------------------------------------------------------------------------
// Stage 1 GEMM: h = silu(X @ Wg) * (X @ Wu)  -> bf16
// Block: 256 rows x 64 cols, loop K=HID in 32-steps. 8 waves, each 32x64 tile.
// row_ids==null => identity rows (shared expert); else gather (routed).
// Staging is two-phase (issue all loads, then convert+store) so the 12 global
// loads per thread stay in flight together instead of wait-to-zero each.
// ---------------------------------------------------------------------------
__global__ __launch_bounds__(256)
void moe_gate_up(const float* __restrict__ x, const int* __restrict__ row_ids,
                 const float* __restrict__ w_gate, const float* __restrict__ w_up,
                 size_t wstride, int nW, bf16* __restrict__ hout) {
  __shared__ bf16 ldsA[256 * 32];
  __shared__ bf16 ldsB1[64 * 32];
  __shared__ bf16 ldsB3[64 * 32];

  const int tid = threadIdx.x;
  const int rb  = blockIdx.x;
  const int n0  = blockIdx.y * 64;
  const int rowBase = rb * 256;
  const float* W1 = w_gate + (size_t)rb * wstride;
  const float* W3 = w_up   + (size_t)rb * wstride;
  const int wid = tid >> 5;

  // hoist row gather: loop-invariant row pointers
  const float* xrow[8];
  {
    const int rl = tid >> 3;
#pragma unroll
    for (int pass = 0; pass < 8; ++pass) {
      const int row = pass * 32 + rl;
      const int rid = row_ids ? row_ids[rowBase + row] : (rowBase + row);
      xrow[pass] = x + (size_t)rid * HID;
    }
  }

  v8f accG[2][4], accU[2][4];
#pragma unroll
  for (int i = 0; i < 2; ++i)
#pragma unroll
    for (int j = 0; j < 4; ++j) { zero8(accG[i][j]); zero8(accU[i][j]); }

  for (int k0 = 0; k0 < HID; k0 += 32) {
    const int rl = tid >> 3;
    const int kk = (tid & 7) * 4;
    const int k  = tid >> 3;
    const int n8 = (tid & 7) * 8;
    const float* p1 = W1 + (size_t)(k0 + k) * nW + n0 + n8;
    const float* p3 = W3 + (size_t)(k0 + k) * nW + n0 + n8;

    // -------- phase 1: issue ALL global loads into registers --------
    float4 xv[8];
#pragma unroll
    for (int pass = 0; pass < 8; ++pass)
      xv[pass] = *(const float4*)(xrow[pass] + k0 + kk);
    float4 w1a = *(const float4*)(p1);
    float4 w1b = *(const float4*)(p1 + 4);
    float4 w3a = *(const float4*)(p3);
    float4 w3b = *(const float4*)(p3 + 4);
    if (k0 + 32 < HID) {  // global_prefetch_b8 of next k-slab
      __builtin_prefetch((const void*)(p1 + (size_t)32 * nW), 0, 1);
      __builtin_prefetch((const void*)(p3 + (size_t)32 * nW), 0, 1);
    }

    // -------- phase 2: convert + store to LDS --------
#pragma unroll
    for (int pass = 0; pass < 8; ++pass) {
      const int row = pass * 32 + rl;
      v4bf h;
      h[0] = (bf16)xv[pass].x; h[1] = (bf16)xv[pass].y;
      h[2] = (bf16)xv[pass].z; h[3] = (bf16)xv[pass].w;
      *(v4bf*)(ldsA + row * 32 + kk) = h;
    }
    {
      float f1[8] = {w1a.x, w1a.y, w1a.z, w1a.w, w1b.x, w1b.y, w1b.z, w1b.w};
      float f3[8] = {w3a.x, w3a.y, w3a.z, w3a.w, w3b.x, w3b.y, w3b.z, w3b.w};
#pragma unroll
      for (int j = 0; j < 8; ++j) {
        ldsB1[(n8 + j) * 32 + k] = (bf16)f1[j];
        ldsB3[(n8 + j) * 32 + k] = (bf16)f3[j];
      }
    }
    __syncthreads();

    const bf16* aBase = ldsA + (wid * 2) * 16 * 32;
    v16bf a0 = load_a_frag(aBase);
    v16bf a1 = load_a_frag(aBase + 16 * 32);
#pragma unroll
    for (int ns = 0; ns < 4; ++ns) {
      v16bf b1 = load_b_frag(ldsB1 + ns * 16 * 32);
      v16bf b3 = load_b_frag(ldsB3 + ns * 16 * 32);
      accG[0][ns] = wmma_bf16(a0, b1, accG[0][ns]);
      accG[1][ns] = wmma_bf16(a1, b1, accG[1][ns]);
      accU[0][ns] = wmma_bf16(a0, b3, accU[0][ns]);
      accU[1][ns] = wmma_bf16(a1, b3, accU[1][ns]);
    }
    __syncthreads();
  }

  // SwiGLU epilogue, store bf16
  const int lane = tid & 31;
  const int ro  = (lane >> 4) * 8;
  const int col = lane & 15;
#pragma unroll
  for (int mt = 0; mt < 2; ++mt) {
#pragma unroll
    for (int ns = 0; ns < 4; ++ns) {
#pragma unroll
      for (int v = 0; v < 8; ++v) {
        const float g = accG[mt][ns][v];
        const float u = accU[mt][ns][v];
        const float s = g / (1.0f + __expf(-g));
        const int row = rowBase + (wid * 2 + mt) * 16 + ro + v;
        hout[(size_t)row * nW + n0 + ns * 16 + col] = (bf16)(s * u);
      }
    }
  }
}

// ---------------------------------------------------------------------------
// Stage 2 GEMM: y = h @ Wd  -> f32 (rows x HID)
// A tile staged with GLOBAL_LOAD_ASYNC_TO_LDS (bypasses VGPRs, ASYNCcnt
// tracked) overlapping the f32->bf16 weight-slab conversion.
// ---------------------------------------------------------------------------
__global__ __launch_bounds__(256)
void moe_down(const bf16* __restrict__ hin, int kdim,
              const float* __restrict__ w_down, size_t wstride,
              float* __restrict__ yout) {
  __shared__ bf16 ldsA[256 * 32];
  __shared__ bf16 ldsB[64 * 32];

  const int tid = threadIdx.x;
  const int rb  = blockIdx.x;
  const int n0  = blockIdx.y * 64;
  const int rowBase = rb * 256;
  const float* W = w_down + (size_t)rb * wstride;
  const int wid = tid >> 5;

  v8f acc[2][4];
#pragma unroll
  for (int i = 0; i < 2; ++i)
#pragma unroll
    for (int j = 0; j < 4; ++j) zero8(acc[i][j]);

  for (int k0 = 0; k0 < kdim; k0 += 32) {
    const int rl = tid >> 3;
    const int kk = (tid & 7) * 4;
    const int k  = tid >> 3;
    const int n8 = (tid & 7) * 8;
    const float* pw = W + (size_t)(k0 + k) * HID + n0 + n8;

    // A tile: pure byte copy -> async DMA directly into LDS (no VGPR round trip)
#pragma unroll
    for (int pass = 0; pass < 8; ++pass) {
      const int row = pass * 32 + rl;
      const bf16* g = hin + (size_t)(rowBase + row) * kdim + k0 + kk;
#if USE_ASYNC_LDS
      async_copy_b64(g, ldsA + row * 32 + kk);
#else
      *(v4bf*)(ldsA + row * 32 + kk) = *(const v4bf*)g;
#endif
    }
    // B slab: two-phase load then convert+transpose-store
    {
      float4 wa = *(const float4*)(pw);
      float4 wb = *(const float4*)(pw + 4);
      if (k0 + 32 < kdim)
        __builtin_prefetch((const void*)(pw + (size_t)32 * HID), 0, 1);
      float f[8] = {wa.x, wa.y, wa.z, wa.w, wb.x, wb.y, wb.z, wb.w};
#pragma unroll
      for (int j = 0; j < 8; ++j) ldsB[(n8 + j) * 32 + k] = (bf16)f[j];
    }
#if USE_ASYNC_LDS
    wait_async0();
#endif
    __syncthreads();

    const bf16* aBase = ldsA + (wid * 2) * 16 * 32;
    v16bf a0 = load_a_frag(aBase);
    v16bf a1 = load_a_frag(aBase + 16 * 32);
#pragma unroll
    for (int ns = 0; ns < 4; ++ns) {
      v16bf b = load_b_frag(ldsB + ns * 16 * 32);
      acc[0][ns] = wmma_bf16(a0, b, acc[0][ns]);
      acc[1][ns] = wmma_bf16(a1, b, acc[1][ns]);
    }
    __syncthreads();
  }

  const int lane = tid & 31;
  const int ro  = (lane >> 4) * 8;
  const int col = lane & 15;
#pragma unroll
  for (int mt = 0; mt < 2; ++mt) {
#pragma unroll
    for (int ns = 0; ns < 4; ++ns) {
#pragma unroll
      for (int v = 0; v < 8; ++v) {
        const int row = rowBase + (wid * 2 + mt) * 16 + ro + v;
        yout[(size_t)row * HID + n0 + ns * 16 + col] = acc[mt][ns][v];
      }
    }
  }
}

// ---------------------------------------------------------------------------
// Combine: out[t] = sum_k w_k * y[slot_k] + sigmoid(x@wsg) * shared[t]
// Token-side gather => deterministic, no atomics.
// ---------------------------------------------------------------------------
__global__ __launch_bounds__(256)
void moe_combine(const float* __restrict__ x, const float* __restrict__ wsg,
                 const int* __restrict__ ridx, const float* __restrict__ rw,
                 const int* __restrict__ epos, const float* __restrict__ ybuf,
                 const float* __restrict__ sbuf, float* __restrict__ out) {
  __shared__ float red[256];
  __shared__ int   se[TOPK];
  __shared__ int   sp[TOPK];
  __shared__ float sw[TOPK];
  const int t = blockIdx.x, tid = threadIdx.x;
  const float* xr = x + (size_t)t * HID;
  float acc = 0.0f;
  for (int i = tid; i < HID; i += 256) acc += xr[i] * wsg[i];
  red[tid] = acc;
  __syncthreads();
  for (int s = 128; s > 0; s >>= 1) {
    if (tid < s) red[tid] += red[tid + s];
    __syncthreads();
  }
  const float sig = 1.0f / (1.0f + __expf(-red[0]));
  if (tid < TOPK) {
    se[tid] = ridx[t * TOPK + tid];
    sp[tid] = epos[t * TOPK + tid];
    sw[tid] = rw[t * TOPK + tid];
  }
  __syncthreads();
  for (int c = tid; c < HID; c += 256) {
    float o = sig * sbuf[(size_t)t * HID + c];
#pragma unroll
    for (int k = 0; k < TOPK; ++k) {
      if (sp[k] >= 0)
        o += sw[k] * ybuf[((size_t)se[k] * CAP + sp[k]) * HID + c];
    }
    out[(size_t)t * HID + c] = o;
  }
}

// ---------------------------------------------------------------------------
extern "C" void kernel_launch(void* const* d_in, const int* in_sizes, int n_in,
                              void* d_out, int out_size, void* d_ws, size_t ws_size,
                              hipStream_t stream) {
  const float* x   = (const float*)d_in[0];
  const float* gw  = (const float*)d_in[1];
  const float* w1  = (const float*)d_in[2];
  const float* w3  = (const float*)d_in[3];
  const float* w2  = (const float*)d_in[4];
  const float* ws1 = (const float*)d_in[5];
  const float* ws3 = (const float*)d_in[6];
  const float* ws2 = (const float*)d_in[7];
  const float* wsg = (const float*)d_in[8];
  float* out = (float*)d_out;

  char* p = (char*)d_ws;
  auto alloc = [&](size_t bytes) -> char* {
    char* r = p;
    p += (bytes + 255) & ~(size_t)255;
    return r;
  };
  int*   ridx  = (int*)  alloc((size_t)T_TOK * TOPK * 4);
  float* rw    = (float*)alloc((size_t)T_TOK * TOPK * 4);
  int*   etok  = (int*)  alloc((size_t)NEXP * CAP * 4);
  int*   epos  = (int*)  alloc((size_t)T_TOK * TOPK * 4);
  bf16*  hbuf  = (bf16*) alloc((size_t)NEXP * CAP * IDIM * 2);
  bf16*  hsbuf = (bf16*) alloc((size_t)T_TOK * ISH * 2);
  float* ybuf  = (float*)alloc((size_t)NEXP * CAP * HID * 4);
  float* sbuf  = (float*)alloc((size_t)T_TOK * HID * 4);

  hipLaunchKernelGGL(moe_router, dim3(T_TOK), dim3(64), 0, stream, x, gw, ridx, rw);
  hipLaunchKernelGGL(moe_dispatch, dim3(NEXP), dim3(32), 0, stream, ridx, etok, epos);
  // routed experts: gate/up + SwiGLU (weights touched exactly once)
  hipLaunchKernelGGL(moe_gate_up, dim3(NEXP, IDIM / 64), dim3(256), 0, stream,
                     x, (const int*)etok, w1, w3, (size_t)HID * IDIM, IDIM, hbuf);
  // shared expert: gate/up + SwiGLU (identity rows)
  hipLaunchKernelGGL(moe_gate_up, dim3(T_TOK / 256, ISH / 64), dim3(256), 0, stream,
                     x, (const int*)nullptr, ws1, ws3, (size_t)0, ISH, hsbuf);
  // down projections
  hipLaunchKernelGGL(moe_down, dim3(NEXP, HID / 64), dim3(256), 0, stream,
                     (const bf16*)hbuf, IDIM, w2, (size_t)IDIM * HID, ybuf);
  hipLaunchKernelGGL(moe_down, dim3(T_TOK / 256, HID / 64), dim3(256), 0, stream,
                     (const bf16*)hsbuf, ISH, ws2, (size_t)0, sbuf);
  // combine + sigmoid-gated shared
  hipLaunchKernelGGL(moe_combine, dim3(T_TOK), dim3(256), 0, stream,
                     x, wsg, ridx, rw, epos, ybuf, sbuf, out);
}
